// VLMEncoder_90168543412487
// MI455X (gfx1250) — matrix operand order repaired and checked by reading, not compiled, not run on hardware
//
#include <hip/hip_runtime.h>

typedef _Float16 half_t;
typedef __attribute__((ext_vector_type(16))) _Float16 v16h;
typedef __attribute__((ext_vector_type(8)))  float    v8f;
typedef int v4i_ __attribute__((vector_size(16)));

#define S_   2048
#define D_   2048
#define H_   16
#define DH_  128
#define DFF_ 8192
#define V_   32000
#define R_   8
#define L_   2
#define KC_  256   // K-chunk staged in LDS per GEMM block (divides 2048, 8192, 32000)

// CDNA5 async global->LDS copy (ASYNCcnt-tracked), with safe fallback
#if defined(__has_builtin)
#if __has_builtin(__builtin_amdgcn_global_load_async_to_lds_b128) && \
    __has_builtin(__builtin_amdgcn_s_wait_asynccnt)
#define USE_ASYNC_LDS 1
#endif
#endif
#ifndef USE_ASYNC_LDS
#define USE_ASYNC_LDS 0
#endif

#if USE_ASYNC_LDS
typedef __attribute__((address_space(1))) v4i_* gas_v4p;
typedef __attribute__((address_space(3))) v4i_* las_v4p;
#endif

__device__ __constant__ float c_nf4[16] = {
  -1.0f, -0.6961928009986877f, -0.5250730514526367f, -0.39491748809814453f,
  -0.28444138169288635f, -0.18477343022823334f, -0.09105003625154495f, 0.0f,
  0.07958029955625534f, 0.16093020141124725f, 0.24611230194568634f,
  0.33791524171829224f, 0.44070982933044434f, 0.5626170039176941f,
  0.7229568362236023f, 1.0f };

union V16u { v16h h; float4 f[2]; };

__device__ __forceinline__ v8f wmma_f16(v16h a, v16h b, v8f c) {
  return __builtin_amdgcn_wmma_f32_16x16x32_f16(false, a, false, b,
                                                (short)0, c, false, false);
}

// ---------------- elementwise / setup kernels ----------------

__global__ __launch_bounds__(256) void k_f32_to_f16(const float* __restrict__ in,
                                                    half_t* __restrict__ out,
                                                    const size_t n) {
  size_t i = (size_t)blockIdx.x * 256 + threadIdx.x;
  if (i < n) out[i] = (half_t)in[i];
}

// out[c][r] = in[r][c], Rr x Cc both multiples of 32
__global__ __launch_bounds__(256) void k_transpose16(const half_t* __restrict__ in,
                                                     half_t* __restrict__ out,
                                                     const int Rr, const int Cc) {
  __shared__ half_t tile[32][33];
  const int bc = blockIdx.x * 32, br = blockIdx.y * 32;
  const int tx = threadIdx.x & 31, ty = threadIdx.x >> 5;  // ty in 0..7
#pragma unroll
  for (int i = 0; i < 4; ++i)
    tile[ty + i * 8][tx] = in[(size_t)(br + ty + i * 8) * Cc + bc + tx];
  __syncthreads();
#pragma unroll
  for (int i = 0; i < 4; ++i)
    out[(size_t)(bc + ty + i * 8) * Rr + br + tx] = tile[tx][ty + i * 8];
}

__global__ __launch_bounds__(256) void k_embed(const int* __restrict__ ids,
                                               const float* __restrict__ emb,
                                               float* __restrict__ h) {
  const int s = blockIdx.x;
  const int id = ids[s];
  for (int d = threadIdx.x; d < D_; d += 256)
    h[(size_t)s * D_ + d] = emb[(size_t)id * D_ + d];
}

__global__ __launch_bounds__(256) void k_rmsnorm(const float* __restrict__ h,
                                                 const float* __restrict__ w,
                                                 half_t* __restrict__ xn) {
  __shared__ float red[8];
  const int s = blockIdx.x, tid = threadIdx.x;
  float acc = 0.f;
  for (int d = tid; d < D_; d += 256) {
    float v = h[(size_t)s * D_ + d];
    acc += v * v;
  }
  for (int off = 16; off; off >>= 1) acc += __shfl_xor(acc, off, 32);
  if ((tid & 31) == 0) red[tid >> 5] = acc;
  __syncthreads();
  float total = 0.f;
#pragma unroll
  for (int i = 0; i < 8; ++i) total += red[i];
  const float rms = rsqrtf(total / (float)D_ + 1e-6f);
  for (int d = tid; d < D_; d += 256)
    xn[(size_t)s * D_ + d] = (half_t)(h[(size_t)s * D_ + d] * rms * w[d]);
}

// NF4 dequant: w16[row,col] = NF4[idx] * scale[row, col/64]
__global__ __launch_bounds__(256) void k_dequant(const int* __restrict__ idx,
                                                 const float* __restrict__ scale,
                                                 half_t* __restrict__ w16,
                                                 const int Kdim, const size_t total) {
  size_t e = (size_t)blockIdx.x * 256 + threadIdx.x;
  if (e >= total) return;
  const size_t row = e / (size_t)Kdim;
  const int col = (int)(e - row * (size_t)Kdim);
  w16[e] = (half_t)(c_nf4[idx[e] & 15] * scale[row * (size_t)(Kdim >> 6) + (col >> 6)]);
}

__global__ __launch_bounds__(256) void k_silu_mul(half_t* __restrict__ gate,
                                                  const half_t* __restrict__ up,
                                                  const size_t n) {
  size_t i = (size_t)blockIdx.x * 256 + threadIdx.x;
  if (i >= n) return;
  float g = (float)gate[i];
  float a = g / (1.f + __expf(-g)) * (float)up[i];
  gate[i] = (half_t)a;
}

// ---------------- WMMA GEMM: y[M=2048, N] = A[M,K] @ W[N,K]^T ----------------
// Block tile 64(M) x 256(N); each wave: 4 M-tiles x 2 N-tiles (8 accumulators).
// A staged in double-buffered LDS (async global->LDS when available).
// mode 0: store f16; mode 1: store f32; mode 2: add into f32 (residual)
__global__ __launch_bounds__(256) void k_gemm(const half_t* __restrict__ A,
                                              const half_t* __restrict__ W,
                                              void* __restrict__ outp,
                                              const int N, const int K,
                                              const int mode) {
  __shared__ half_t As[2][64][KC_];
  const int tid = threadIdx.x;
  const int wave = tid >> 5, lane = tid & 31;
  const int hb = lane >> 4, ln = lane & 15;
  const int m0 = blockIdx.y * 64;
  const int nw = blockIdx.x * 256 + wave * 32;   // two 16-wide N tiles per wave

  // staging map: 4 threads per row, 8 float4 (128B) contiguous per thread
  const int srow = tid >> 2;
  const int sq   = tid & 3;

  v8f acc[4][2];
#pragma unroll
  for (int mt = 0; mt < 4; ++mt)
#pragma unroll
    for (int t = 0; t < 2; ++t)
      acc[mt][t] = (v8f){0.f, 0.f, 0.f, 0.f, 0.f, 0.f, 0.f, 0.f};

  const half_t* wrow0 = W + (size_t)(nw + ln) * K;
  const half_t* wrow1 = W + (size_t)(nw + 16 + ln) * K;

  auto stage = [&](int buf, int k0) {
    const float4* src =
        reinterpret_cast<const float4*>(A + (size_t)(m0 + srow) * K + k0) + sq * 8;
    float4* dst = reinterpret_cast<float4*>(&As[buf][srow][0]) + sq * 8;
#if USE_ASYNC_LDS
#pragma unroll
    for (int i = 0; i < 8; ++i)
      __builtin_amdgcn_global_load_async_to_lds_b128(
          (gas_v4p)(src + i), (las_v4p)(dst + i), 0, 0);
#else
    float4 tmp[8];
#pragma unroll
    for (int i = 0; i < 8; ++i) tmp[i] = src[i];
#pragma unroll
    for (int i = 0; i < 8; ++i) dst[i] = tmp[i];
#endif
  };

  const int nchunks = K / KC_;
  stage(0, 0);
#if USE_ASYNC_LDS
  __builtin_amdgcn_s_wait_asynccnt(0);
#endif
  __syncthreads();

  for (int c = 0; c < nchunks; ++c) {
    const int buf = c & 1;
    const int k0 = c * KC_;
    if (c + 1 < nchunks) {
      stage(buf ^ 1, k0 + KC_);
      __builtin_prefetch(wrow0 + k0 + KC_, 0, 1);
    }
#pragma unroll
    for (int kk = 0; kk < KC_; kk += 32) {
      V16u b0, b1;
      const float4* wp0 = reinterpret_cast<const float4*>(wrow0 + k0 + kk + hb * 16);
      const float4* wp1 = reinterpret_cast<const float4*>(wrow1 + k0 + kk + hb * 16);
      b0.f[0] = wp0[0];
      b0.f[1] = wp0[1];
      b1.f[0] = wp1[0];
      b1.f[1] = wp1[1];
#pragma unroll
      for (int mt = 0; mt < 4; ++mt) {
        V16u a;
        a.f[0] = *reinterpret_cast<const float4*>(&As[buf][mt * 16 + ln][kk + hb * 16]);
        a.f[1] = *reinterpret_cast<const float4*>(&As[buf][mt * 16 + ln][kk + hb * 16 + 8]);
        acc[mt][0] = wmma_f16(a.h, b0.h, acc[mt][0]);
        acc[mt][1] = wmma_f16(a.h, b1.h, acc[mt][1]);
      }
    }
#if USE_ASYNC_LDS
    __builtin_amdgcn_s_wait_asynccnt(0);
#endif
    __syncthreads();
  }

  if (mode == 0) {
    half_t* out = (half_t*)outp;
#pragma unroll
    for (int mt = 0; mt < 4; ++mt)
#pragma unroll
      for (int t = 0; t < 2; ++t)
#pragma unroll
        for (int r = 0; r < 8; ++r)
          out[(size_t)(m0 + mt * 16 + 8 * hb + r) * N + nw + t * 16 + ln] =
              (half_t)acc[mt][t][r];
  } else if (mode == 1) {
    float* out = (float*)outp;
#pragma unroll
    for (int mt = 0; mt < 4; ++mt)
#pragma unroll
      for (int t = 0; t < 2; ++t)
#pragma unroll
        for (int r = 0; r < 8; ++r)
          out[(size_t)(m0 + mt * 16 + 8 * hb + r) * N + nw + t * 16 + ln] =
              acc[mt][t][r];
  } else {
    float* out = (float*)outp;
#pragma unroll
    for (int mt = 0; mt < 4; ++mt)
#pragma unroll
      for (int t = 0; t < 2; ++t)
#pragma unroll
        for (int r = 0; r < 8; ++r) {
          size_t i = (size_t)(m0 + mt * 16 + 8 * hb + r) * N + nw + t * 16 + ln;
          out[i] += acc[mt][t][r];
        }
  }
}

// ---------------- LoRA: t1 = x @ A^T ; y += (t1 @ B^T) * 2.0 ----------------

__global__ __launch_bounds__(256) void k_lora1(const half_t* __restrict__ x,
                                               const float* __restrict__ Am,
                                               float* __restrict__ t1) {
  const int wave = threadIdx.x >> 5, lane = threadIdx.x & 31;
  const int s = blockIdx.x * 8 + wave;
  float acc[R_];
#pragma unroll
  for (int r = 0; r < R_; ++r) acc[r] = 0.f;
  for (int d = lane; d < D_; d += 32) {
    const float xv = (float)x[(size_t)s * D_ + d];
#pragma unroll
    for (int r = 0; r < R_; ++r) acc[r] += xv * Am[(size_t)r * D_ + d];
  }
#pragma unroll
  for (int r = 0; r < R_; ++r) {
    float v = acc[r];
    for (int off = 16; off; off >>= 1) v += __shfl_xor(v, off, 32);
    if (lane == 0) t1[(size_t)s * R_ + r] = v;
  }
}

// fmode 0: add into f16 buffer; fmode 1: add into f32 buffer
__global__ __launch_bounds__(256) void k_lora2(const float* __restrict__ t1,
                                               const float* __restrict__ Bm,
                                               void* __restrict__ yp,
                                               const int fmode) {
  const size_t idx = (size_t)blockIdx.x * 256 + threadIdx.x;
  const int s = (int)(idx / D_);
  const int d = (int)(idx - (size_t)s * D_);
  float v = 0.f;
#pragma unroll
  for (int r = 0; r < R_; ++r)
    v += t1[(size_t)s * R_ + r] * Bm[(size_t)d * R_ + r];
  v *= 2.0f;  // LORA_SCALE = 16/8
  if (fmode == 0) {
    half_t* y = (half_t*)yp;
    y[idx] = (half_t)((float)y[idx] + v);
  } else {
    float* y = (float*)yp;
    y[idx] += v;
  }
}

// ---------------- Flash attention (one wave = one head x 16-query tile) -----

__global__ __launch_bounds__(32) void k_attn(const half_t* __restrict__ q,
                                             const half_t* __restrict__ kmat,
                                             const half_t* __restrict__ vT,
                                             half_t* __restrict__ o) {
  __shared__ half_t Pl[16][32];
  const int lane = threadIdx.x;
  const int hb = lane >> 4, ln = lane & 15;
  const int m0 = blockIdx.x * 16;
  const int head = blockIdx.y;
  const float scale = 0.08838834764831845f;  // 1/sqrt(128)
  const v8f vzero = {0.f, 0.f, 0.f, 0.f, 0.f, 0.f, 0.f, 0.f};

  // Q tile in A layout: 4 chunks of K=32 along DH
  V16u qa[4];
#pragma unroll
  for (int c = 0; c < 4; ++c) {
    const half_t* p = q + ((size_t)(m0 + ln) * H_ + head) * DH_ + c * 32 + hb * 16;
    qa[c].f[0] = *reinterpret_cast<const float4*>(p);
    qa[c].f[1] = *reinterpret_cast<const float4*>(p + 8);
  }

  v8f oa[8];
#pragma unroll
  for (int t = 0; t < 8; ++t) oa[t] = vzero;
  float mrow[8], lsum[8];
#pragma unroll
  for (int r = 0; r < 8; ++r) { mrow[r] = -3.0e30f; lsum[r] = 0.f; }

  const int ktiles = (m0 + 16 + 31) / 32;
  for (int kt = 0; kt < ktiles; ++kt) {
    const int kbase = kt * 32;
    v8f c0 = vzero, c1 = vzero;
#pragma unroll
    for (int c = 0; c < 4; ++c) {
      V16u b0, b1;
      const half_t* p0 =
          kmat + ((size_t)(kbase + ln) * H_ + head) * DH_ + c * 32 + hb * 16;
      const half_t* p1 =
          kmat + ((size_t)(kbase + 16 + ln) * H_ + head) * DH_ + c * 32 + hb * 16;
      b0.f[0] = *reinterpret_cast<const float4*>(p0);
      b0.f[1] = *reinterpret_cast<const float4*>(p0 + 8);
      b1.f[0] = *reinterpret_cast<const float4*>(p1);
      b1.f[1] = *reinterpret_cast<const float4*>(p1 + 8);
      c0 = wmma_f16(qa[c].h, b0.h, c0);
      c1 = wmma_f16(qa[c].h, b1.h, c1);
    }
    // online softmax update (rows m = m0 + 8*hb + r; keys n = kbase + ln [+16])
    float p0v[8], p1v[8];
#pragma unroll
    for (int r = 0; r < 8; ++r) {
      const int qrow = m0 + 8 * hb + r;
      float s0 = c0[r] * scale;
      float s1 = c1[r] * scale;
      if (kbase + ln > qrow) s0 = -3.0e30f;
      if (kbase + 16 + ln > qrow) s1 = -3.0e30f;
      float mx = fmaxf(s0, s1);
      for (int off = 8; off; off >>= 1) mx = fmaxf(mx, __shfl_xor(mx, off, 16));
      const float nm = fmaxf(mrow[r], mx);
      const float corr = __expf(mrow[r] - nm);
      const float e0 = __expf(s0 - nm);
      const float e1 = __expf(s1 - nm);
      float ts = e0 + e1;
      for (int off = 8; off; off >>= 1) ts += __shfl_xor(ts, off, 16);
      lsum[r] = lsum[r] * corr + ts;
      mrow[r] = nm;
      p0v[r] = e0;
      p1v[r] = e1;
#pragma unroll
      for (int t = 0; t < 8; ++t) oa[t][r] *= corr;
    }
    // re-layout P: C-layout -> LDS -> A-layout (16x32)
    __syncthreads();
#pragma unroll
    for (int r = 0; r < 8; ++r) {
      Pl[8 * hb + r][ln] = (half_t)p0v[r];
      Pl[8 * hb + r][16 + ln] = (half_t)p1v[r];
    }
    __syncthreads();
    V16u pa;
    pa.f[0] = *reinterpret_cast<const float4*>(&Pl[ln][hb * 16]);
    pa.f[1] = *reinterpret_cast<const float4*>(&Pl[ln][hb * 16 + 8]);
    // O += P @ V  (B from pre-transposed vT[D][S], contiguous along keys)
#pragma unroll
    for (int t = 0; t < 8; ++t) {
      V16u vb;
      const half_t* vp =
          vT + (size_t)(head * DH_ + t * 16 + ln) * S_ + kbase + hb * 16;
      vb.f[0] = *reinterpret_cast<const float4*>(vp);
      vb.f[1] = *reinterpret_cast<const float4*>(vp + 8);
      oa[t] = wmma_f16(pa.h, vb.h, oa[t]);
    }
  }
#pragma unroll
  for (int t = 0; t < 8; ++t)
#pragma unroll
    for (int r = 0; r < 8; ++r)
      o[(size_t)(m0 + 8 * hb + r) * D_ + head * DH_ + t * 16 + ln] =
          (half_t)(oa[t][r] / lsum[r]);
}

// ---------------- gumbel-softmax over V ----------------

__device__ __forceinline__ float gumbel_z(float lg, float uv) {
  float gn = -logf(-logf(uv + 1e-10f) + 1e-10f);
  return lg + gn;  // TAU = 1
}

__global__ __launch_bounds__(256) void k_rowmax(const float* __restrict__ logits,
                                                const float* __restrict__ u,
                                                float* __restrict__ rmax) {
  __shared__ float red[8];
  const int s = blockIdx.x, tid = threadIdx.x;
  float mx = -3.0e30f;
  for (int v = tid; v < V_; v += 256) {
    size_t i = (size_t)s * V_ + v;
    mx = fmaxf(mx, gumbel_z(logits[i], u[i]));
  }
  for (int off = 16; off; off >>= 1) mx = fmaxf(mx, __shfl_xor(mx, off, 32));
  if ((tid & 31) == 0) red[tid >> 5] = mx;
  __syncthreads();
  if (tid == 0) {
    float t = red[0];
#pragma unroll
    for (int i = 1; i < 8; ++i) t = fmaxf(t, red[i]);
    rmax[s] = t;
  }
}

__global__ __launch_bounds__(256) void k_expwrite(const float* __restrict__ logits,
                                                  const float* __restrict__ u,
                                                  const float* __restrict__ rmax,
                                                  half_t* __restrict__ g,
                                                  float* __restrict__ rsum) {
  __shared__ float red[8];
  const int s = blockIdx.x, tid = threadIdx.x;
  const float m = rmax[s];
  float acc = 0.f;
  for (int v = tid; v < V_; v += 256) {
    size_t i = (size_t)s * V_ + v;
    float e = __expf(gumbel_z(logits[i], u[i]) - m);
    g[i] = (half_t)e;
    acc += e;
  }
  for (int off = 16; off; off >>= 1) acc += __shfl_xor(acc, off, 32);
  if ((tid & 31) == 0) red[tid >> 5] = acc;
  __syncthreads();
  if (tid == 0) {
    float t = 0.f;
#pragma unroll
    for (int i = 0; i < 8; ++i) t += red[i];
    rsum[s] = t;
  }
}

__global__ __launch_bounds__(256) void k_gscale(half_t* __restrict__ g,
                                                const float* __restrict__ rsum) {
  size_t i = (size_t)blockIdx.x * 256 + threadIdx.x;
  const int s = (int)(i / V_);
  g[i] = (half_t)((float)g[i] / rsum[s]);
}

// ---------------- host orchestration ----------------

extern "C" void kernel_launch(void* const* d_in, const int* in_sizes, int n_in,
                              void* d_out, int out_size, void* d_ws, size_t ws_size,
                              hipStream_t stream) {
  (void)in_sizes; (void)n_in; (void)out_size; (void)ws_size;

  const int*   ids        = (const int*)  d_in[0];
  const float* u          = (const float*)d_in[1];
  const float* emb        = (const float*)d_in[2];
  const float* attn_scale = (const float*)d_in[3];
  const float* gu_scale   = (const float*)d_in[4];
  const float* dn_scale   = (const float*)d_in[5];
  const float* lora_A     = (const float*)d_in[6];
  const float* lora_B     = (const float*)d_in[7];
  const float* norm_w     = (const float*)d_in[8];
  const float* fnorm_w    = (const float*)d_in[9];
  const int*   attn_idx   = (const int*)  d_in[10];
  const int*   gu_idx     = (const int*)  d_in[11];
  const int*   dn_idx     = (const int*)  d_in[12];

  char* ws = (char*)d_ws;
  size_t off = 0;
  float*  h     = (float*) (ws + off); off += (size_t)S_ * D_ * 4;
  half_t* xn    = (half_t*)(ws + off); off += (size_t)S_ * D_ * 2;
  half_t* qb    = (half_t*)(ws + off); off += (size_t)S_ * D_ * 2;
  half_t* kb    = (half_t*)(ws + off); off += (size_t)S_ * D_ * 2;
  half_t* vb    = (half_t*)(ws + off); off += (size_t)S_ * D_ * 2;
  half_t* vT    = (half_t*)(ws + off); off += (size_t)S_ * D_ * 2;
  half_t* ob    = (half_t*)(ws + off); off += (size_t)S_ * D_ * 2;
  float*  t1    = (float*) (ws + off); off += (size_t)S_ * R_ * 4;
  float*  rmax  = (float*) (ws + off); off += (size_t)S_ * 4;
  float*  rsum  = (float*) (ws + off); off += (size_t)S_ * 4;
  half_t* wbuf  = (half_t*)(ws + off); off += (size_t)2 * DFF_ * D_ * 2;
  half_t* gate  = (half_t*)(ws + off); off += (size_t)S_ * DFF_ * 2;
  half_t* up    = (half_t*)(ws + off); off += (size_t)S_ * DFF_ * 2;
  half_t* emb16 = (half_t*)(ws + off); off += (size_t)V_ * D_ * 2;
  half_t* embT  = (half_t*)(ws + off); off += (size_t)V_ * D_ * 2;
  half_t* g16   = emb16;  // overlay: emb16 dead after the logits GEMM

  float* soft   = (float*)d_out;                       // [S, D]
  float* logits = (float*)d_out + (size_t)S_ * D_;     // [S, V]

  // setup: f16 embedding (+transpose) and token gather
  k_f32_to_f16<<<(unsigned)(((size_t)V_ * D_) / 256), 256, 0, stream>>>(
      emb, emb16, (size_t)V_ * D_);
  k_transpose16<<<dim3(D_ / 32, V_ / 32), 256, 0, stream>>>(emb16, embT, V_, D_);
  k_embed<<<S_, 256, 0, stream>>>(ids, emb, h);

  const dim3 gD(D_ / 256, S_ / 64);
  const dim3 gF(DFF_ / 256, S_ / 64);
  const dim3 gV(V_ / 256, S_ / 64);

  for (int l = 0; l < L_; ++l) {
    // ---- attention block ----
    k_rmsnorm<<<S_, 256, 0, stream>>>(h, norm_w + (size_t)(l * 2 + 0) * D_, xn);
    k_dequant<<<(unsigned)(((size_t)4 * D_ * D_) / 256), 256, 0, stream>>>(
        attn_idx + (size_t)l * 4 * D_ * D_,
        attn_scale + (size_t)l * 4 * D_ * (D_ / 64),
        wbuf, D_, (size_t)4 * D_ * D_);
    half_t* qkv[3] = {qb, kb, vb};
    for (int i = 0; i < 3; ++i) {
      k_gemm<<<gD, 256, 0, stream>>>(xn, wbuf + (size_t)i * D_ * D_, qkv[i],
                                     D_, D_, 0);
      k_lora1<<<S_ / 8, 256, 0, stream>>>(
          xn, lora_A + (size_t)(l * 4 + i) * R_ * D_, t1);
      k_lora2<<<(unsigned)(((size_t)S_ * D_) / 256), 256, 0, stream>>>(
          t1, lora_B + (size_t)(l * 4 + i) * D_ * R_, qkv[i], 0);
    }
    k_transpose16<<<dim3(D_ / 32, S_ / 32), 256, 0, stream>>>(vb, vT, S_, D_);
    k_attn<<<dim3(S_ / 16, H_), 32, 0, stream>>>(qb, kb, vT, ob);
    k_gemm<<<gD, 256, 0, stream>>>(ob, wbuf + (size_t)3 * D_ * D_, h, D_, D_, 2);
    k_lora1<<<S_ / 8, 256, 0, stream>>>(
        ob, lora_A + (size_t)(l * 4 + 3) * R_ * D_, t1);
    k_lora2<<<(unsigned)(((size_t)S_ * D_) / 256), 256, 0, stream>>>(
        t1, lora_B + (size_t)(l * 4 + 3) * D_ * R_, h, 1);

    // ---- FFN block ----
    k_rmsnorm<<<S_, 256, 0, stream>>>(h, norm_w + (size_t)(l * 2 + 1) * D_, xn);
    k_dequant<<<(unsigned)(((size_t)2 * DFF_ * D_) / 256), 256, 0, stream>>>(
        gu_idx + (size_t)l * 2 * DFF_ * D_,
        gu_scale + (size_t)l * 2 * DFF_ * (D_ / 64),
        wbuf, D_, (size_t)2 * DFF_ * D_);
    k_gemm<<<gF, 256, 0, stream>>>(xn, wbuf, gate, DFF_, D_, 0);
    k_gemm<<<gF, 256, 0, stream>>>(xn, wbuf + (size_t)DFF_ * D_, up, DFF_, D_, 0);
    k_silu_mul<<<(unsigned)(((size_t)S_ * DFF_) / 256), 256, 0, stream>>>(
        gate, up, (size_t)S_ * DFF_);
    k_dequant<<<(unsigned)(((size_t)D_ * DFF_) / 256), 256, 0, stream>>>(
        dn_idx + (size_t)l * D_ * DFF_, dn_scale + (size_t)l * D_ * (DFF_ / 64),
        wbuf, DFF_, (size_t)D_ * DFF_);
    k_gemm<<<gD, 256, 0, stream>>>(gate, wbuf, h, D_, DFF_, 2);
  }

  // ---- head: final norm, logits, gumbel-softmax, soft embeddings ----
  k_rmsnorm<<<S_, 256, 0, stream>>>(h, fnorm_w, xn);
  k_gemm<<<gV, 256, 0, stream>>>(xn, emb16, logits, V_, D_, 1);
  k_rowmax<<<S_, 256, 0, stream>>>(logits, u, rmax);
  k_expwrite<<<S_, 256, 0, stream>>>(logits, u, rmax, g16, rsum);
  k_gscale<<<(unsigned)(((size_t)S_ * V_) / 256), 256, 0, stream>>>(g16, rsum);
  k_gemm<<<gD, 256, 0, stream>>>(g16, embT, soft, D_, V_, 1);
}
